// OnsetsDetector_9002251453026
// MI455X (gfx1250) — matrix operand order repaired and checked by reading, not compiled, not run on hardware
//
#include <hip/hip_runtime.h>
#include <math.h>

#define NHASH 8
#define HEADS 8
#define DHEAD 64
#define DIM 512
#define DEPTH 8
#define SPECD 229
#define NPITCH 88
#define TSEQ 2048
#define BATCH 2
#define MTOK (BATCH*TSEQ)       /* 4096 */
#define BKT 64
#define NBKT (TSEQ/BKT)         /* 32 */
#define BHD (BATCH*HEADS)       /* 16 */
#define KCONV 768
#define SELF_VAL -5.0e4f

typedef _Float16 v16h __attribute__((ext_vector_type(16)));
typedef _Float16 v8h  __attribute__((ext_vector_type(8)));
typedef float    v8f  __attribute__((ext_vector_type(8)));

__device__ inline v8f zero8f() {
  v8f z;
  #pragma unroll
  for (int i = 0; i < 8; ++i) z[i] = 0.0f;
  return z;
}

// A fragment (16x32 f16, M across lanes, K split 0-7/16-23 vs 8-15/24-31 by lane half)
__device__ inline v16h load_fragA(const _Float16* base, int ld, int lane) {
  const _Float16* p = base + (size_t)(lane & 15) * ld + ((lane >> 4) << 3);
  v8h lo = *(const v8h*)(p);
  v8h hi = *(const v8h*)(p + 16);
  v16h r;
  #pragma unroll
  for (int i = 0; i < 8; ++i) { r[i] = lo[i]; r[i + 8] = hi[i]; }
  return r;
}

// B fragment (32x16 f16): lane = N column, K split 0-7/16-23 vs 8-15/24-31 by lane half
__device__ inline v16h load_fragB(const _Float16* base, int ld, int lane) {
  const _Float16* p = base + (size_t)(lane & 15) * ld + ((lane >> 4) << 3);
  v8h lo = *(const v8h*)(p);
  v8h hi = *(const v8h*)(p + 16);
  v16h r;
  #pragma unroll
  for (int i = 0; i < 8; ++i) { r[i] = lo[i]; r[i + 8] = hi[i]; }
  return r;
}

__device__ inline v8f wmma32(v16h a, v16h b, v8f c) {
  return __builtin_amdgcn_wmma_f32_16x16x32_f16(false, a, false, b, (short)0, c, false, false);
}

// ---------------------------------------------------------------------------
// Generic NT GEMM: C[M,Npad] = epi(A[M,K] @ Bt[Npad,K]^T)
// MODE 0: Cf = acc
// MODE 1: Cf = resid + acc + bias            (residual update, f32)
// MODE 2: Ch = gelu(acc + bias)              (f16 out)
// MODE 3: Cf[m*Nreal+n] = acc + bias, n<Nreal (compact masked store)
// MODE 4: Cf = relu(acc + bias)
// ---------------------------------------------------------------------------
template <int MODE>
__global__ __launch_bounds__(256) void gemm_nt(
    const _Float16* __restrict__ A, const _Float16* __restrict__ Bt,
    const float* __restrict__ bias, const float* __restrict__ resid,
    float* __restrict__ Cf, _Float16* __restrict__ Ch,
    int M, int Npad, int K, int Nreal)
{
  const int lane = threadIdx.x & 31;
  const int wave = threadIdx.x >> 5;
  const int wm = wave & 3;           // 4 waves along M
  const int wn = wave >> 2;          // 2 waves along N
  const int tm = blockIdx.x * 128 + wm * 32;
  const int tn = blockIdx.y * 128 + wn * 64;

  v8f acc[2][4];
  #pragma unroll
  for (int i = 0; i < 2; ++i)
    #pragma unroll
    for (int j = 0; j < 4; ++j) acc[i][j] = zero8f();

  for (int k0 = 0; k0 < K; k0 += 32) {
    if (k0 + 32 < K) {
      __builtin_prefetch(A + (size_t)tm * K + k0 + 32, 0, 1);
      __builtin_prefetch(Bt + (size_t)tn * K + k0 + 32, 0, 1);
    }
    v16h a0 = load_fragA(A + (size_t)(tm)      * K + k0, K, lane);
    v16h a1 = load_fragA(A + (size_t)(tm + 16) * K + k0, K, lane);
    v16h b0 = load_fragB(Bt + (size_t)(tn)      * K + k0, K, lane);
    v16h b1 = load_fragB(Bt + (size_t)(tn + 16) * K + k0, K, lane);
    v16h b2 = load_fragB(Bt + (size_t)(tn + 32) * K + k0, K, lane);
    v16h b3 = load_fragB(Bt + (size_t)(tn + 48) * K + k0, K, lane);
    acc[0][0] = wmma32(a0, b0, acc[0][0]);
    acc[0][1] = wmma32(a0, b1, acc[0][1]);
    acc[0][2] = wmma32(a0, b2, acc[0][2]);
    acc[0][3] = wmma32(a0, b3, acc[0][3]);
    acc[1][0] = wmma32(a1, b0, acc[1][0]);
    acc[1][1] = wmma32(a1, b1, acc[1][1]);
    acc[1][2] = wmma32(a1, b2, acc[1][2]);
    acc[1][3] = wmma32(a1, b3, acc[1][3]);
  }

  const int rbase = (lane >> 4) << 3;   // 0 or 8
  #pragma unroll
  for (int i = 0; i < 2; ++i) {
    #pragma unroll
    for (int j = 0; j < 4; ++j) {
      int n = tn + j * 16 + (lane & 15);
      #pragma unroll
      for (int r = 0; r < 8; ++r) {
        int m = tm + i * 16 + rbase + r;
        float v = acc[i][j][r];
        if (MODE == 0) {
          Cf[(size_t)m * Npad + n] = v;
        } else if (MODE == 1) {
          size_t idx = (size_t)m * Npad + n;
          Cf[idx] = resid[idx] + v + bias[n];
        } else if (MODE == 2) {
          float x = v + bias[n];
          float g = 0.5f * x * (1.0f + erff(x * 0.70710678f));
          Ch[(size_t)m * Npad + n] = (_Float16)g;
        } else if (MODE == 3) {
          if (n < Nreal) Cf[(size_t)m * Nreal + n] = v + bias[n];
        } else { // 4: relu
          float x = v + bias[n];
          Cf[(size_t)m * Npad + n] = x > 0.0f ? x : 0.0f;
        }
      }
    }
  }
}

// ---------------------------------------------------------------------------
// Weight prep: fp32 (L,K,N) -> f16 K-major (L,Npad,K), zero-padded cols
// ---------------------------------------------------------------------------
__global__ void cvt_transpose(const float* __restrict__ W, _Float16* __restrict__ Wt,
                              int K, int N, int Npad, int L)
{
  size_t i = (size_t)blockIdx.x * 256 + threadIdx.x;
  size_t total = (size_t)L * Npad * K;
  if (i >= total) return;
  int k = (int)(i % K);
  size_t t = i / K;
  int n = (int)(t % Npad);
  int l = (int)(t / Npad);
  _Float16 v = (_Float16)0.0f;
  if (n < N) v = (_Float16)W[((size_t)l * K + k) * N + n];
  Wt[i] = v;
}

__global__ void conv_wt_kernel(const float* __restrict__ cw, _Float16* __restrict__ wt)
{
  size_t i = (size_t)blockIdx.x * 256 + threadIdx.x;
  if (i >= (size_t)DIM * KCONV) return;
  int o = (int)(i / KCONV);
  int kp = (int)(i % KCONV);
  int kk = kp >> 8, c = kp & 255;
  _Float16 v = (_Float16)0.0f;
  if (c < SPECD) v = (_Float16)cw[((size_t)o * SPECD + c) * 3 + kk];
  wt[i] = v;
}

__global__ void im2col_kernel(const float* __restrict__ spec, _Float16* __restrict__ X)
{
  size_t i = (size_t)blockIdx.x * 256 + threadIdx.x;
  if (i >= (size_t)MTOK * KCONV) return;
  int r = (int)(i / KCONV);
  int kp = (int)(i % KCONV);
  int b = r >> 11, t = r & 2047;
  int kk = kp >> 8, c = kp & 255;
  int tt = t + kk - 1;
  _Float16 v = (_Float16)0.0f;
  if (c < SPECD && tt >= 0 && tt < TSEQ)
    v = (_Float16)spec[((size_t)(b * TSEQ + tt)) * SPECD + c];
  X[i] = v;
}

__global__ void pe_add_kernel(const float* __restrict__ conv,
                              const float* __restrict__ per, const float* __restrict__ pec,
                              float* __restrict__ x1, float* __restrict__ x2)
{
  size_t i = (size_t)blockIdx.x * 256 + threadIdx.x;
  if (i >= (size_t)MTOK * DIM) return;
  int c = (int)(i & 511);
  int tg = (int)(i >> 9);
  int t = tg & 2047;
  float p = (c < 256) ? per[(size_t)(t >> 6) * 256 + c] : pec[(size_t)(t & 63) * 256 + (c - 256)];
  float v = conv[i] + p;
  x1[i] = v; x2[i] = v;
}

__global__ __launch_bounds__(128) void ln_kernel(const float* __restrict__ x,
                                                 const float* __restrict__ g,
                                                 const float* __restrict__ b,
                                                 _Float16* __restrict__ y)
{
  int row = blockIdx.x;
  const float* xr = x + (size_t)row * DIM;
  __shared__ float s1[4], s2[4];
  float sum = 0.0f, sq = 0.0f;
  for (int c = threadIdx.x; c < DIM; c += 128) { float v = xr[c]; sum += v; sq += v * v; }
  #pragma unroll
  for (int s = 1; s < 32; s <<= 1) { sum += __shfl_xor(sum, s, 32); sq += __shfl_xor(sq, s, 32); }
  int w = threadIdx.x >> 5;
  if ((threadIdx.x & 31) == 0) { s1[w] = sum; s2[w] = sq; }
  __syncthreads();
  sum = s1[0] + s1[1] + s1[2] + s1[3];
  sq  = s2[0] + s2[1] + s2[2] + s2[3];
  float mean = sum * (1.0f / DIM);
  float var = sq * (1.0f / DIM) - mean * mean;
  float rstd = rsqrtf(var + 1e-5f);
  for (int c = threadIdx.x; c < DIM; c += 128)
    y[(size_t)row * DIM + c] = (_Float16)((xr[c] - mean) * rstd * g[c] + b[c]);
}

// one thread per (bh, nh, t): 16 rotations of the 64-d qk, argmax over [rot, -rot]
__global__ void hash_kernel(const float* __restrict__ qk, const float* __restrict__ rot,
                            int* __restrict__ buckets)
{
  size_t i = (size_t)blockIdx.x * 256 + threadIdx.x;
  if (i >= (size_t)BHD * NHASH * TSEQ) return;
  int t = (int)(i & 2047);
  int nh = (int)((i >> 11) & 7);
  int bh = (int)(i >> 14);
  int b = bh >> 3, h = bh & 7;
  const float* q = qk + ((size_t)(b * TSEQ + t)) * DIM + h * DHEAD;
  float acc[16];
  #pragma unroll
  for (int j = 0; j < 16; ++j) acc[j] = 0.0f;
  for (int d = 0; d < DHEAD; ++d) {
    float qd = q[d];
    const float* rr = rot + ((size_t)d * NHASH + nh) * 16;
    #pragma unroll
    for (int j = 0; j < 16; ++j) acc[j] += qd * rr[j];
  }
  int best = 0;
  float bv = acc[0];
  #pragma unroll
  for (int j = 1; j < 16; ++j) if (acc[j] > bv) { bv = acc[j]; best = j; }
  #pragma unroll
  for (int j = 0; j < 16; ++j) { float nv = -acc[j]; if (nv > bv) { bv = nv; best = 16 + j; } }
  buckets[(size_t)(bh * NHASH + nh) * TSEQ + t] = best;
}

// stable counting sort: one wave per (bh, nh), one lane per bucket value
__global__ __launch_bounds__(32) void sort_kernel(const int* __restrict__ buckets,
                                                  int* __restrict__ st)
{
  int grp = blockIdx.x;               // bh*NHASH + nh
  int v = threadIdx.x;                // bucket value 0..31
  const int* bk = buckets + (size_t)grp * TSEQ;
  int cnt = 0;
  for (int t = 0; t < TSEQ; ++t) cnt += (bk[t] == v);
  int x = cnt;
  #pragma unroll
  for (int s = 1; s < 32; s <<= 1) { int y = __shfl_up(x, s, 32); if (v >= s) x += y; }
  int off = x - cnt;                  // exclusive prefix
  int* out = st + (size_t)grp * TSEQ;
  for (int t = 0; t < TSEQ; ++t)
    if (bk[t] == v) out[off++] = t;
}

// ---------------------------------------------------------------------------
// Chunked LSH attention: one workgroup per (bh, nh, chunk); WMMA on LDS tiles
// ---------------------------------------------------------------------------
__global__ __launch_bounds__(128) void attn_chunk(
    const float* __restrict__ qk, const float* __restrict__ vv,
    const int* __restrict__ st, _Float16* __restrict__ obuf, float* __restrict__ lg)
{
  __shared__ __align__(32) _Float16 qs[64 * 64];
  __shared__ __align__(32) _Float16 ks[128 * 64];
  __shared__ __align__(32) _Float16 vt[64 * 128];
  __shared__ __align__(32) _Float16 ps[64 * 128];
  __shared__ int tpos[128];

  const int ch = blockIdx.x, nh = blockIdx.y, bh = blockIdx.z;
  const int j = threadIdx.x;
  const int lane = threadIdx.x & 31;
  const int w = threadIdx.x >> 5;
  const int b = bh >> 3, h = bh & 7;
  const size_t grp = (size_t)(bh * NHASH + nh);

  // gather: 64 q tokens (= first 64 kv) + 64 prev-chunk kv
  {
    int srcCh = (j < 64) ? ch : (ch + NBKT - 1) % NBKT;
    int pos = st[grp * TSEQ + srcCh * BKT + (j & 63)];
    tpos[j] = pos;
    const float* qrow = qk + ((size_t)(b * TSEQ + pos)) * DIM + h * DHEAD;
    const float* vrow = vv + ((size_t)(b * TSEQ + pos)) * DIM + h * DHEAD;
    float n2 = 0.0f;
    for (int d = 0; d < DHEAD; ++d) { float q = qrow[d]; n2 += q * q; }
    float rn = rsqrtf(fmaxf(n2, 1e-24f));
    for (int d = 0; d < DHEAD; ++d) {
      float q = qrow[d];
      ks[j * DHEAD + d] = (_Float16)(q * rn);
      if (j < BKT) qs[j * DHEAD + d] = (_Float16)(q * 0.125f);  // * DH^-0.5
      vt[d * 128 + j] = (_Float16)vrow[d];
    }
  }
  __syncthreads();

  // dots (64x128): wave w owns rows [w*16, w*16+16)
  v8f dacc[8];
  #pragma unroll
  for (int n = 0; n < 8; ++n) dacc[n] = zero8f();
  #pragma unroll
  for (int k0 = 0; k0 < 64; k0 += 32) {
    v16h a = load_fragA(qs + (size_t)(w * 16) * DHEAD + k0, DHEAD, lane);
    #pragma unroll
    for (int n = 0; n < 8; ++n) {
      v16h bb = load_fragB(ks + (size_t)(n * 16) * DHEAD + k0, DHEAD, lane);
      dacc[n] = wmma32(a, bb, dacc[n]);
    }
  }

  // mask + logsumexp softmax per row, write p (f16) to LDS
  const int rbase = (lane >> 4) << 3;
  const int ncol = lane & 15;
  #pragma unroll
  for (int r = 0; r < 8; ++r) {
    int m = w * 16 + rbase + r;
    int qp = tpos[m];
    float mx = -1e30f;
    #pragma unroll
    for (int n = 0; n < 8; ++n) {
      int col = n * 16 + ncol;
      float d = dacc[n][r];
      if (tpos[col] == qp) d = SELF_VAL;
      dacc[n][r] = d;
      mx = fmaxf(mx, d);
    }
    #pragma unroll
    for (int s = 1; s < 16; s <<= 1) mx = fmaxf(mx, __shfl_xor(mx, s, 32));
    float sm = 0.0f;
    #pragma unroll
    for (int n = 0; n < 8; ++n) sm += expf(dacc[n][r] - mx);
    #pragma unroll
    for (int s = 1; s < 16; s <<= 1) sm += __shfl_xor(sm, s, 32);
    float inv = 1.0f / sm;
    #pragma unroll
    for (int n = 0; n < 8; ++n) {
      int col = n * 16 + ncol;
      ps[m * 128 + col] = (_Float16)(expf(dacc[n][r] - mx) * inv);
    }
    if (ncol == 0) lg[grp * TSEQ + qp] = mx + logf(sm);
  }
  __syncthreads();

  // out (64x64) = p (64x128) @ v (128x64), Bt = vt (64 x 128 K-major)
  v8f oacc[4];
  #pragma unroll
  for (int n = 0; n < 4; ++n) oacc[n] = zero8f();
  #pragma unroll
  for (int k0 = 0; k0 < 128; k0 += 32) {
    v16h a = load_fragA(ps + (size_t)(w * 16) * 128 + k0, 128, lane);
    #pragma unroll
    for (int n = 0; n < 4; ++n) {
      v16h bb = load_fragB(vt + (size_t)(n * 16) * 128 + k0, 128, lane);
      oacc[n] = wmma32(a, bb, oacc[n]);
    }
  }
  // scatter to unsorted positions
  #pragma unroll
  for (int n = 0; n < 4; ++n) {
    int d = n * 16 + ncol;
    #pragma unroll
    for (int r = 0; r < 8; ++r) {
      int m = w * 16 + rbase + r;
      obuf[(grp * TSEQ + tpos[m]) * DHEAD + d] = (_Float16)oacc[n][r];
    }
  }
}

// softmax-merge over hash rounds, reorder heads -> (b,t,512) f16
__global__ void combine_kernel(const _Float16* __restrict__ obuf, const float* __restrict__ lg,
                               _Float16* __restrict__ attn)
{
  size_t i = (size_t)blockIdx.x * 256 + threadIdx.x;
  if (i >= (size_t)BHD * TSEQ * DHEAD) return;
  int d = (int)(i & 63);
  int t = (int)((i >> 6) & 2047);
  int bh = (int)(i >> 17);
  int b = bh >> 3, h = bh & 7;
  float mx = -1e30f;
  #pragma unroll
  for (int nh = 0; nh < NHASH; ++nh)
    mx = fmaxf(mx, lg[(size_t)(bh * NHASH + nh) * TSEQ + t]);
  float s = 0.0f, acc = 0.0f;
  #pragma unroll
  for (int nh = 0; nh < NHASH; ++nh) {
    size_t g = (size_t)(bh * NHASH + nh) * TSEQ + t;
    float wgt = expf(lg[g] - mx);
    s += wgt;
    acc += wgt * (float)obuf[g * DHEAD + d];
  }
  attn[((size_t)(b * TSEQ + t)) * DIM + h * DHEAD + d] = (_Float16)(acc / s);
}

__global__ void avg_kernel(const float* __restrict__ x1, const float* __restrict__ x2,
                           _Float16* __restrict__ xm)
{
  size_t i = (size_t)blockIdx.x * 256 + threadIdx.x;
  if (i >= (size_t)MTOK * DIM) return;
  xm[i] = (_Float16)(0.5f * (x1[i] + x2[i]));
}

// ---------------------------------------------------------------------------
extern "C" void kernel_launch(void* const* d_in, const int* in_sizes, int n_in,
                              void* d_out, int out_size, void* d_ws, size_t ws_size,
                              hipStream_t stream)
{
  (void)in_sizes; (void)n_in; (void)out_size; (void)ws_size;
  const float* spec   = (const float*)d_in[0];
  const float* conv_w = (const float*)d_in[1];
  const float* conv_b = (const float*)d_in[2];
  const float* pe_row = (const float*)d_in[3];
  const float* pe_col = (const float*)d_in[4];
  const float* lnA_g  = (const float*)d_in[5];
  const float* lnA_b  = (const float*)d_in[6];
  const float* wqk    = (const float*)d_in[7];
  const float* wv     = (const float*)d_in[8];
  const float* wo     = (const float*)d_in[9];
  const float* bo     = (const float*)d_in[10];
  const float* lnF_g  = (const float*)d_in[11];
  const float* lnF_b  = (const float*)d_in[12];
  const float* w1     = (const float*)d_in[13];
  const float* b1     = (const float*)d_in[14];
  const float* w2     = (const float*)d_in[15];
  const float* b2     = (const float*)d_in[16];
  const float* lin_w  = (const float*)d_in[17];
  const float* lin_b  = (const float*)d_in[18];
  const float* rot    = (const float*)d_in[19];

  char* ws = (char*)d_ws;
  size_t off = 0;
  auto arena = [&](size_t bytes) -> char* {
    off = (off + 255) & ~(size_t)255;
    char* p = ws + off;
    off += bytes;
    return p;
  };

  _Float16* wt_qk  = (_Float16*)arena((size_t)DEPTH * 512 * 512 * 2);
  _Float16* wt_v   = (_Float16*)arena((size_t)DEPTH * 512 * 512 * 2);
  _Float16* wt_o   = (_Float16*)arena((size_t)DEPTH * 512 * 512 * 2);
  _Float16* wt_w1  = (_Float16*)arena((size_t)DEPTH * 2048 * 512 * 2);
  _Float16* wt_w2  = (_Float16*)arena((size_t)DEPTH * 512 * 2048 * 2);
  _Float16* wt_cv  = (_Float16*)arena((size_t)DIM * KCONV * 2);
  _Float16* wt_lin = (_Float16*)arena((size_t)128 * 512 * 2);
  _Float16* xcol   = (_Float16*)arena((size_t)MTOK * KCONV * 2);
  float*    x1     = (float*)   arena((size_t)MTOK * DIM * 4);
  float*    x2     = (float*)   arena((size_t)MTOK * DIM * 4);
  _Float16* lnbuf  = (_Float16*)arena((size_t)MTOK * DIM * 2);
  float*    qkbuf  = (float*)   arena((size_t)MTOK * DIM * 4);
  float*    vbuf   = (float*)   arena((size_t)MTOK * DIM * 4);
  int*      bkbuf  = (int*)     arena((size_t)BHD * NHASH * TSEQ * 4);
  int*      stbuf  = (int*)     arena((size_t)BHD * NHASH * TSEQ * 4);
  _Float16* obuf   = (_Float16*)arena((size_t)BHD * NHASH * TSEQ * DHEAD * 2);
  float*    lgbuf  = (float*)   arena((size_t)BHD * NHASH * TSEQ * 4);
  _Float16* attnb  = (_Float16*)arena((size_t)MTOK * DIM * 2);
  _Float16* h1buf  = (_Float16*)arena((size_t)MTOK * 2048 * 2);
  _Float16* xm     = (_Float16*)arena((size_t)MTOK * DIM * 2);

  auto blocks = [](size_t total) { return (unsigned)((total + 255) / 256); };

  // ---- weight prep (deterministic, every call) ----
  cvt_transpose<<<blocks((size_t)DEPTH*512*512), 256, 0, stream>>>(wqk,  wt_qk, 512, 512, 512, DEPTH);
  cvt_transpose<<<blocks((size_t)DEPTH*512*512), 256, 0, stream>>>(wv,   wt_v,  512, 512, 512, DEPTH);
  cvt_transpose<<<blocks((size_t)DEPTH*512*512), 256, 0, stream>>>(wo,   wt_o,  512, 512, 512, DEPTH);
  cvt_transpose<<<blocks((size_t)DEPTH*2048*512), 256, 0, stream>>>(w1,  wt_w1, 512, 2048, 2048, DEPTH);
  cvt_transpose<<<blocks((size_t)DEPTH*512*2048), 256, 0, stream>>>(w2,  wt_w2, 2048, 512, 512, DEPTH);
  cvt_transpose<<<blocks((size_t)128*512), 256, 0, stream>>>(lin_w, wt_lin, 512, NPITCH, 128, 1);
  conv_wt_kernel<<<blocks((size_t)DIM*KCONV), 256, 0, stream>>>(conv_w, wt_cv);

  // ---- frontend: im2col conv GEMM (relu+bias) + axial PE, duplicate streams ----
  im2col_kernel<<<blocks((size_t)MTOK*KCONV), 256, 0, stream>>>(spec, xcol);
  gemm_nt<4><<<dim3(MTOK/128, DIM/128), 256, 0, stream>>>(
      xcol, wt_cv, conv_b, nullptr, qkbuf, nullptr, MTOK, DIM, KCONV, DIM);
  pe_add_kernel<<<blocks((size_t)MTOK*DIM), 256, 0, stream>>>(qkbuf, pe_row, pe_col, x1, x2);

  // ---- reversible Reformer layers ----
  for (int l = 0; l < DEPTH; ++l) {
    const _Float16* wqk_l = wt_qk + (size_t)l * 512 * 512;
    const _Float16* wv_l  = wt_v  + (size_t)l * 512 * 512;
    const _Float16* wo_l  = wt_o  + (size_t)l * 512 * 512;
    const _Float16* w1_l  = wt_w1 + (size_t)l * 2048 * 512;
    const _Float16* w2_l  = wt_w2 + (size_t)l * 512 * 2048;
    const float* rot_l = rot + (size_t)l * DHEAD * NHASH * 16;

    // attention branch: x1 += LSH(LN(x2))
    ln_kernel<<<MTOK, 128, 0, stream>>>(x2, lnA_g + l * DIM, lnA_b + l * DIM, lnbuf);
    gemm_nt<0><<<dim3(MTOK/128, DIM/128), 256, 0, stream>>>(
        lnbuf, wqk_l, nullptr, nullptr, qkbuf, nullptr, MTOK, DIM, DIM, DIM);
    gemm_nt<0><<<dim3(MTOK/128, DIM/128), 256, 0, stream>>>(
        lnbuf, wv_l, nullptr, nullptr, vbuf, nullptr, MTOK, DIM, DIM, DIM);
    hash_kernel<<<blocks((size_t)BHD*NHASH*TSEQ), 256, 0, stream>>>(qkbuf, rot_l, bkbuf);
    sort_kernel<<<BHD * NHASH, 32, 0, stream>>>(bkbuf, stbuf);
    attn_chunk<<<dim3(NBKT, NHASH, BHD), 128, 0, stream>>>(qkbuf, vbuf, stbuf, obuf, lgbuf);
    combine_kernel<<<blocks((size_t)BHD*TSEQ*DHEAD), 256, 0, stream>>>(obuf, lgbuf, attnb);
    gemm_nt<1><<<dim3(MTOK/128, DIM/128), 256, 0, stream>>>(
        attnb, wo_l, bo + l * DIM, x1, x1, nullptr, MTOK, DIM, DIM, DIM);

    // FFN branch: x2 += W2(gelu(W1(LN(x1))))
    ln_kernel<<<MTOK, 128, 0, stream>>>(x1, lnF_g + l * DIM, lnF_b + l * DIM, lnbuf);
    gemm_nt<2><<<dim3(MTOK/128, 2048/128), 256, 0, stream>>>(
        lnbuf, w1_l, b1 + l * 2048, nullptr, nullptr, h1buf, MTOK, 2048, DIM, 2048);
    gemm_nt<1><<<dim3(MTOK/128, DIM/128), 256, 0, stream>>>(
        h1buf, w2_l, b2 + l * DIM, x2, x2, nullptr, MTOK, DIM, 2048, DIM);
  }

  // ---- head: y = 0.5*(x1+x2) @ lin_w + lin_b ----
  avg_kernel<<<blocks((size_t)MTOK*DIM), 256, 0, stream>>>(x1, x2, xm);
  gemm_nt<3><<<dim3(MTOK/128, 1), 256, 0, stream>>>(
      xm, wt_lin, lin_b, nullptr, (float*)d_out, nullptr, MTOK, 128, DIM, NPITCH);
}
